// Vision_Encoder_64183991271743
// MI455X (gfx1250) — matrix operand order repaired and checked by reading, not compiled, not run on hardware
//
// Spikformer-style SNN vision encoder for MI455X (gfx1250), wave32 + WMMA.
// conv1..4 + all transformer linears + attention run on v_wmma_f32_16x16x32_f16
// (binary spikes / small-int scores are exact in f16). B tiles are staged in
// LDS, double-buffered via global_load_async_to_lds_b128 (ASYNCcnt) when the
// toolchain exposes it. conv0/LIF/BN/pool are bandwidth-bound fp32 kernels.
#include <hip/hip_runtime.h>
#include <math.h>

typedef __attribute__((ext_vector_type(16))) _Float16 v16h;
typedef __attribute__((ext_vector_type(8)))  _Float16 v8h;
typedef __attribute__((ext_vector_type(8)))  float    v8f;
typedef __attribute__((ext_vector_type(4)))  int      v4i;

static constexpr int T_STEPS = 4;
static constexpr int EMB     = 384;
static constexpr int HEADS   = 12;
static constexpr int HDIM    = 32;
static constexpr int NB      = 8;
static constexpr int TBATCH  = T_STEPS * NB;   // 32
static constexpr int NTOK    = 576;
static constexpr int HIDDEN  = EMB * 4;

#if defined(__has_builtin)
#if __has_builtin(__builtin_amdgcn_global_load_async_to_lds_b128)
#define HAVE_ASYNC 1
#endif
#endif
#ifndef HAVE_ASYNC
#define HAVE_ASYNC 0
#endif

// ---------------------------------------------------------------------------
// WMMA fragment loaders (wave32, blockDim.x == 32).
// A (16x32 f16, MxK): lane l -> row m=l&15; elems 0..7 <-> K=(l>>4)*8+0..7,
//                     elems 8..15 <-> K=16+(l>>4)*8+0..7.
// B (32x16 f16, KxN): lane l -> col n=l&15; elems 0..7 <-> K=(l>>4)*8+0..7,
//                     elems 8..15 <-> K=16+(l>>4)*8+0..7.
// C/D (16x16 f32):    lane l -> col n=l&15; VGPR r -> row m=r+8*(l>>4).
// ---------------------------------------------------------------------------
__device__ __forceinline__ v16h fragA(const _Float16* p, int ld) {
  int lane = threadIdx.x & 31;
  const _Float16* r = p + (size_t)(lane & 15) * ld + ((lane >> 4) * 8);
  v8h lo = *(const v8h*)(r);
  v8h hi = *(const v8h*)(r + 16);
  v16h a;
#pragma unroll
  for (int i = 0; i < 8; ++i) { a[i] = lo[i]; a[8 + i] = hi[i]; }
  return a;
}

__device__ __forceinline__ v16h fragB(const _Float16* p, int ld) {
  int lane = threadIdx.x & 31;
  const _Float16* r = p + (size_t)(lane & 15) * ld + ((lane >> 4) * 8);
  v8h lo = *(const v8h*)(r);
  v8h hi = *(const v8h*)(r + 16);
  v16h b;
#pragma unroll
  for (int i = 0; i < 8; ++i) { b[i] = lo[i]; b[8 + i] = hi[i]; }
  return b;
}

__device__ __forceinline__ v8f wmma_f16(v16h a, v16h b, v8f c) {
  return __builtin_amdgcn_wmma_f32_16x16x32_f16(false, a, false, b, (short)0, c,
                                                false, false);
}

__device__ __forceinline__ void wait_async_all() {
#if HAVE_ASYNC
#if __has_builtin(__builtin_amdgcn_s_wait_asynccnt)
  __builtin_amdgcn_s_wait_asynccnt(0);
#else
  asm volatile("s_wait_asynccnt 0x0" ::: "memory");
#endif
#endif
}

// Cooperative fill of a 32-token x 32-k f16 tile (rows pitch 32 halfs).
// 128 16B chunks spread over 96 threads; async (ASYNCcnt) when available.
__device__ __forceinline__ void fill_tileB(const _Float16* Bg, int K, int k0,
                                           _Float16* dst, int tid) {
  for (int c = tid; c < 128; c += 96) {
    int row = c >> 2, seg = (c & 3) * 8;
    const _Float16* g = Bg + (size_t)row * K + k0 + seg;
    _Float16* l = dst + row * 32 + seg;
#if HAVE_ASYNC
    typedef __attribute__((address_space(1))) v4i as1_v4i;
    typedef __attribute__((address_space(3))) v4i as3_v4i;
    __builtin_amdgcn_global_load_async_to_lds_b128(
        (as1_v4i*)(void*)const_cast<_Float16*>(g), (as3_v4i*)(void*)l, 0, 0);
#else
    *(v8h*)l = *(const v8h*)g;
#endif
  }
}

// ---------------------------------------------------------------------------
// Token-major GEMM with LDS-staged B and 2x2 register blocking per wave:
//   out[b][n][o] = scale[o]*(sum_c W[o][c]*act[b][n][c]) + shift[o] (+resid)
// W:[M,K] f16, act:[batch,N,K] f16, out:[batch,N,M] f32.
// grid = (M/96, N/32, batch), block = (32,3): 3 waves, each 32(M)x32(N).
// All dims divide tiles exactly -> no predication, EXEC all-ones for WMMA.
// ---------------------------------------------------------------------------
__global__ void gemm_tok(const _Float16* __restrict__ W,
                         const _Float16* __restrict__ act,
                         float* __restrict__ out,
                         const float* __restrict__ resid,
                         const float* __restrict__ scale,
                         const float* __restrict__ shift,
                         int M, int N, int K) {
  __shared__ __align__(16) _Float16 sB[2][32 * 32];
  int lane = threadIdx.x & 31;
  int wv = threadIdx.y;
  int tid = wv * 32 + lane;
  int mbase = blockIdx.x * 96 + wv * 32;
  int nbase = blockIdx.y * 32;
  int b = blockIdx.z;
  const _Float16* A0 = W + (size_t)mbase * K;
  const _Float16* Bg = act + ((size_t)b * N + nbase) * K;

  fill_tileB(Bg, K, 0, sB[0], tid);

  v8f acc00 = {}, acc01 = {}, acc10 = {}, acc11 = {};
  int nk = K / 32;
  for (int i = 0; i < nk; ++i) {
    int buf = i & 1;
    wait_async_all();
    __syncthreads();                 // tile i visible; prev reads of buf^1 done
    if (i + 1 < nk) fill_tileB(Bg, K, (i + 1) * 32, sB[buf ^ 1], tid);
    if (i + 4 < nk) {                // prefetch the weight stream
      const _Float16* pf = A0 + (size_t)(lane & 15) * K + (size_t)(i + 4) * 32;
      __builtin_prefetch(pf, 0, 0);
      __builtin_prefetch(pf + (size_t)16 * K, 0, 0);
    }
    v16h a0 = fragA(A0 + (size_t)i * 32, K);
    v16h a1 = fragA(A0 + (size_t)16 * K + (size_t)i * 32, K);
    v16h b0 = fragB(&sB[buf][0], 32);
    v16h b1 = fragB(&sB[buf][16 * 32], 32);
    acc00 = wmma_f16(a0, b0, acc00);
    acc01 = wmma_f16(a0, b1, acc01);
    acc10 = wmma_f16(a1, b0, acc10);
    acc11 = wmma_f16(a1, b1, acc11);
  }

  int tok0 = nbase + (lane & 15);
  int mr0 = mbase + (lane >> 4) * 8;
  v8f accs[4] = {acc00, acc01, acc10, acc11};
#pragma unroll
  for (int mi = 0; mi < 2; ++mi) {
#pragma unroll
    for (int ni = 0; ni < 2; ++ni) {
      v8f a = accs[mi * 2 + ni];
      int tok = tok0 + ni * 16;
      int mr = mr0 + mi * 16;
      float* op = out + ((size_t)b * N + tok) * M + mr;
      const float* rp = resid ? resid + ((size_t)b * N + tok) * M + mr : nullptr;
#pragma unroll
      for (int r = 0; r < 8; ++r) {
        float zv = a[r] * scale[mr + r] + shift[mr + r];
        if (rp) zv += rp[r];
        op[r] = zv;
      }
    }
  }
}

// ---------------------------------------------------------------------------
// Fused binary-spike attention per (t,b,head): o = (q k^T) v * 0.125
// q,k: [TB][N][C] f16 spikes; vT: [TB][H][HDIM][N] f16 spikes.
// One wave per 16-token x 32-d tile; q A-fragment register-resident; f32
// scores (exact small ints) re-laid via LDS as the f16 A-fragment of GEMM 2.
// grid = (N/16, 1, TB*HEADS), block = 32.
// ---------------------------------------------------------------------------
__global__ void attn_fused(const _Float16* __restrict__ qh,
                           const _Float16* __restrict__ kh,
                           const _Float16* __restrict__ vT,
                           float* __restrict__ out,
                           int N, int C) {
  constexpr int PITCH = 40;
  __shared__ __align__(16) _Float16 sS[16 * PITCH];

  int nbase = blockIdx.x * 16;
  int tb = blockIdx.z / HEADS;
  int h  = blockIdx.z % HEADS;
  int lane = threadIdx.x & 31;

  const _Float16* qp = qh + ((size_t)tb * N + nbase) * C + h * HDIM;
  const _Float16* kp = kh + (size_t)tb * N * C + h * HDIM;
  const _Float16* vp = vT + ((size_t)tb * HEADS + h) * (size_t)HDIM * N;

  v16h qa = fragA(qp, C);
  v8f acc0 = {}, acc1 = {};

  for (int m0 = 0; m0 < N; m0 += 32) {
    v16h kb0 = fragB(kp + (size_t)m0 * C, C);
    v16h kb1 = fragB(kp + (size_t)(m0 + 16) * C, C);
    v8f s0 = {}, s1 = {};
    s0 = wmma_f16(qa, kb0, s0);
    s1 = wmma_f16(qa, kb1, s1);

    int col = lane & 15, rbase = (lane >> 4) * 8;
#pragma unroll
    for (int r = 0; r < 8; ++r) {
      sS[(rbase + r) * PITCH + col]      = (_Float16)s0[r];
      sS[(rbase + r) * PITCH + 16 + col] = (_Float16)s1[r];
    }
    __syncthreads();

    v16h a2 = fragA(&sS[0], PITCH);
    v16h bv0 = fragB(vp + m0, N);
    v16h bv1 = fragB(vp + (size_t)16 * N + m0, N);
    acc0 = wmma_f16(a2, bv0, acc0);
    acc1 = wmma_f16(a2, bv1, acc1);
    __syncthreads();
  }

  int cch = h * HDIM + (lane & 15);
#pragma unroll
  for (int r = 0; r < 8; ++r) {
    int tok = nbase + (lane >> 4) * 8 + r;
    float* op = out + ((size_t)tb * N + tok) * C + cch;
    op[0]  = acc0[r] * 0.125f;
    op[16] = acc1[r] * 0.125f;
  }
}

// ---------------------------------------------------------------------------
// Elementwise / stencil kernels
// ---------------------------------------------------------------------------
__global__ void prep_bn(const float* __restrict__ g, const float* __restrict__ b,
                        const float* __restrict__ m, const float* __restrict__ v,
                        const float* __restrict__ bias,
                        float* __restrict__ scale, float* __restrict__ shift, int C) {
  int c = blockIdx.x * blockDim.x + threadIdx.x;
  if (c >= C) return;
  float s = g[c] * rsqrtf(v[c] + 1e-5f);
  float bi = bias ? bias[c] : 0.f;
  scale[c] = s;
  shift[c] = (bi - m[c]) * s + b[c];
}

// fp32 [M,K] -> f16 [M,Kpad] (zero pad K..Kpad)
__global__ void cvt_f16_pad(const float* __restrict__ w, _Float16* __restrict__ o,
                            int M, int K, int Kpad) {
  size_t idx = (size_t)blockIdx.x * blockDim.x + threadIdx.x;
  if (idx >= (size_t)M * Kpad) return;
  int k = (int)(idx % Kpad);
  int m = (int)(idx / Kpad);
  o[idx] = (k < K) ? (_Float16)w[(size_t)m * K + k] : (_Float16)0.f;
}

// LIF: v += (x - v)/2 ; spike = (v >= vth) ; hard reset; T sequential.
__global__ void lif_f16(const float* __restrict__ x, _Float16* __restrict__ s,
                        int S, float vth) {
  int i = blockIdx.x * blockDim.x + threadIdx.x;
  if (i >= S) return;
  float v = 0.f;
  for (int t = 0; t < T_STEPS; ++t) {
    float xv = x[(size_t)t * S + i];
    v += (xv - v) * 0.5f;
    float sp = (v >= vth) ? 1.f : 0.f;
    s[(size_t)t * S + i] = (_Float16)sp;
    v *= (1.f - sp);
  }
}

// LIF over T with channel-first fp32 input (constant across T) ->
// channel-last f16 spikes [T][B][HW][C].
__global__ void lif_cf_to_cl(const float* __restrict__ x, _Float16* __restrict__ s,
                             int Bn, int C, int HW, float vth) {
  size_t idx = (size_t)blockIdx.x * blockDim.x + threadIdx.x;
  if (idx >= (size_t)Bn * HW * C) return;
  int c = (int)(idx % C);
  size_t t2 = idx / C;
  int hw = (int)(t2 % HW);
  int b = (int)(t2 / HW);
  float xv = x[((size_t)b * C + c) * HW + hw];
  float v = 0.f;
  for (int t = 0; t < T_STEPS; ++t) {
    v += (xv - v) * 0.5f;
    float sp = (v >= vth) ? 1.f : 0.f;
    s[(((size_t)(t * Bn + b)) * HW + hw) * C + c] = (_Float16)sp;
    v *= (1.f - sp);
  }
}

// Direct conv3x3 (+fused BN) fp32, channel-first (only conv0: Cin=3).
__global__ void conv3x3_bn(const float* __restrict__ in, const float* __restrict__ w,
                           const float* __restrict__ scale, const float* __restrict__ shift,
                           float* __restrict__ out,
                           int NBatch, int Cin, int Cout, int H, int W) {
  size_t idx = (size_t)blockIdx.x * blockDim.x + threadIdx.x;
  size_t total = (size_t)NBatch * Cout * H * W;
  if (idx >= total) return;
  int xw = (int)(idx % W);
  size_t t = idx / W;
  int yh = (int)(t % H); t /= H;
  int oc = (int)(t % Cout);
  int b  = (int)(t / Cout);
  const float* wp = w + (size_t)oc * Cin * 9;
  const float* ip = in + (size_t)b * Cin * H * W;
  float acc = 0.f;
  for (int ic = 0; ic < Cin; ++ic) {
    const float* pp = ip + (size_t)ic * H * W;
    const float* ww = wp + ic * 9;
#pragma unroll
    for (int ky = 0; ky < 3; ++ky) {
      int y = yh + ky - 1;
      if (y < 0 || y >= H) continue;
#pragma unroll
      for (int kx = 0; kx < 3; ++kx) {
        int x = xw + kx - 1;
        if (x < 0 || x >= W) continue;
        acc += pp[(size_t)y * W + x] * ww[ky * 3 + kx];
      }
    }
  }
  out[idx] = acc * scale[oc] + shift[oc];
}

// im2col for 3x3 pad1 on channel-last f16 spikes: one image.
// in:[H*W][Cin] -> col:[H*W][Kpad], k = ic*9 + ky*3 + kx (zero pad to Kpad).
__global__ void im2col3x3(const _Float16* __restrict__ in, _Float16* __restrict__ col,
                          int H, int W, int Cin, int Kpad) {
  size_t idx = (size_t)blockIdx.x * blockDim.x + threadIdx.x;
  size_t total = (size_t)H * W * Kpad;
  if (idx >= total) return;
  int k = (int)(idx % Kpad);
  int hw = (int)(idx / Kpad);
  _Float16 v = (_Float16)0.f;
  if (k < Cin * 9) {
    int ic = k / 9, r = k % 9;
    int y = hw / W + r / 3 - 1;
    int x = hw % W + r % 3 - 1;
    if (y >= 0 && y < H && x >= 0 && x < W)
      v = in[((size_t)y * W + x) * Cin + ic];
  }
  col[idx] = v;
}

// MaxPool2d(k=3,s=2,p=1), channel-last fp32: [img][H][W][C] -> [img][H/2][W/2][C]
__global__ void maxpool3s2_cl(const float* __restrict__ in, float* __restrict__ out,
                              int NImg, int H, int W, int C) {
  int Ho = H >> 1, Wo = W >> 1;
  size_t idx = (size_t)blockIdx.x * blockDim.x + threadIdx.x;
  size_t total = (size_t)NImg * Ho * Wo * C;
  if (idx >= total) return;
  int c = (int)(idx % C);
  size_t t = idx / C;
  int xo = (int)(t % Wo); t /= Wo;
  int yo = (int)(t % Ho);
  int img = (int)(t / Ho);
  const float* ip = in + (size_t)img * H * W * C;
  float mx = -INFINITY;
  for (int dy = -1; dy <= 1; ++dy) {
    int y = 2 * yo + dy;
    if (y < 0 || y >= H) continue;
    for (int dx = -1; dx <= 1; ++dx) {
      int x = 2 * xo + dx;
      if (x < 0 || x >= W) continue;
      float vv = ip[((size_t)y * W + x) * C + c];
      mx = vv > mx ? vv : mx;
    }
  }
  out[idx] = mx;
}

// v spikes [TB][N][C] f16 -> vT [TB][H][HDIM][N] f16
__global__ void v_to_headT(const _Float16* __restrict__ in, _Float16* __restrict__ out,
                           int NBatch, int N, int C) {
  size_t idx = (size_t)blockIdx.x * blockDim.x + threadIdx.x;
  if (idx >= (size_t)NBatch * N * C) return;
  int c = (int)(idx % C);
  size_t t = idx / C;
  int n = (int)(t % N);
  int b = (int)(t / N);
  int h = c >> 5, d = c & 31;
  out[(((size_t)b * HEADS + h) * HDIM + d) * N + n] = in[idx];
}

__global__ void head_reduce(const float* __restrict__ x, const float* __restrict__ wT,
                            float* __restrict__ z, int Bn, int N, int C) {
  int i = blockIdx.x * blockDim.x + threadIdx.x;
  if (i >= Bn * C) return;
  int b = i / C, c = i % C;
  float acc = 0.f;
  for (int t = 0; t < T_STEPS; ++t) {
    const float* xp = x + ((size_t)(t * Bn + b) * N) * C + c;
    float s = 0.f;
    for (int n = 0; n < N; ++n) s += xp[(size_t)n * C];
    acc += wT[t] * s;
  }
  z[i] = acc / (float)N;
}

__global__ void final_proj(const float* __restrict__ z, const float* __restrict__ W,
                           const float* __restrict__ bias, float* __restrict__ out,
                           int Bn, int O, int C) {
  int i = blockIdx.x * blockDim.x + threadIdx.x;
  if (i >= Bn * O) return;
  int b = i / O, o = i % O;
  const float* zp = z + (size_t)b * C;
  const float* wp = W + (size_t)o * C;
  float acc = bias[o];
  for (int c = 0; c < C; ++c) acc += zp[c] * wp[c];
  out[i] = acc;
}

// ---------------------------------------------------------------------------
// Host orchestration. Param index map (setup_inputs dict order):
//  0: pixel_values; tokenizer i=0..4: base=1+5i -> {w,g,b,m,v};
//  block j: base=26+33j -> q.w,+1..4 | +5 k.w,+6..9 | +10 v.w,+11..14
//           +15 proj.w,+16 proj.b,+17..20 | +21 m1.w,+22 m1.b,+23..26
//           +27 m2.w,+28 m2.b,+29..32
//  158: wT(4); 159: proj.w [512,384]; 160: proj.b [512]
// ---------------------------------------------------------------------------
extern "C" void kernel_launch(void* const* d_in, const int* in_sizes, int n_in,
                              void* d_out, int out_size, void* d_ws, size_t ws_size,
                              hipStream_t stream) {
  (void)in_sizes; (void)n_in; (void)out_size; (void)ws_size;
  auto P = [&](int i) -> const float* { return (const float*)d_in[i]; };
  const float* pix = P(0);

  char* ws = (char*)d_ws;
  size_t off = 0;
  auto alloc = [&](size_t bytes) -> void* {
    void* p = ws + off;
    off += (bytes + 255) & ~(size_t)255;
    return p;
  };
  const size_t F = sizeof(float);
  float*    RA = (float*)alloc((size_t)3538944 * F);    // y0 / later p3s (f16)
  _Float16* RB = (_Float16*)alloc((size_t)14155776 * 2); // s0_cl / p2s
  float*    RC = (float*)alloc((size_t)28311552 * F);    // y1 / y3 / z_tmp
  _Float16* RD = (_Float16*)alloc((size_t)28311552 * 2); // s1_cl / h_big
  float*    RE = (float*)alloc((size_t)56623104 * F);    // y2 / transformer f16 bufs
  float*    RF = (float*)alloc((size_t)14155776 * F);    // p2 / p3 (fp32)
  float*    RX = (float*)alloc((size_t)7077888 * F);     // master x [TB][N][C]
  _Float16* COL0 = (_Float16*)alloc((size_t)7962624 * 2);
  _Float16* COL1 = (_Float16*)alloc((size_t)7962624 * 2);
  _Float16* WA = (_Float16*)alloc((size_t)1327104 * 2);  // padded f16 weights
  float* SC = (float*)alloc(1536 * F);
  float* SH = (float*)alloc(1536 * F);
  float* FZ = (float*)alloc(3072 * F);

  _Float16* s0  = RB;                 // [TB][9216][48]
  _Float16* s1  = RD;                 // [TB][9216][96]
  _Float16* p2s = RB;                 // [TB][2304][192]
  _Float16* p3s = (_Float16*)RA;      // [TB][576][384]
  float*    z_tmp = RC;               // up to [TB][N][1536]
  _Float16* xl_h = (_Float16*)RE;
  _Float16* q_h  = xl_h + 7077888;
  _Float16* k_h  = q_h + 7077888;
  _Float16* v_h  = k_h + 7077888;
  _Float16* vT_h = v_h + 7077888;
  _Float16* h_big = RD;
  _Float16* o_h  = q_h;               // reuse (q dead post-attention)
  float*    o_f  = z_tmp;

  auto G = [](size_t n, int t) { return dim3((unsigned)((n + t - 1) / t)); };
  const dim3 GBLK(32, 3);  // gemm_tok block: 3 waves

  // ------- conv layer driver: im2col (per image) + WMMA GEMM, BN fused -----
  auto conv_wmma = [&](const _Float16* spk, int H, int W, int Cin, int Cout,
                       int wIdx, float* yout) {
    int K = Cin * 9;
    int Kpad = (K + 31) & ~31;
    int HW = H * W;
    cvt_f16_pad<<<G((size_t)Cout * Kpad, 256), 256, 0, stream>>>(P(wIdx), WA, Cout, K, Kpad);
    prep_bn<<<G(Cout, 64), 64, 0, stream>>>(P(wIdx + 1), P(wIdx + 2), P(wIdx + 3),
                                            P(wIdx + 4), nullptr, SC, SH, Cout);
    for (int img = 0; img < TBATCH; ++img) {
      _Float16* cb = (img & 1) ? COL1 : COL0;
      im2col3x3<<<G((size_t)HW * Kpad, 256), 256, 0, stream>>>(
          spk + (size_t)img * HW * Cin, cb, H, W, Cin, Kpad);
      gemm_tok<<<dim3(Cout / 96, HW / 32, 1), GBLK, 0, stream>>>(
          WA, cb, yout + (size_t)img * HW * Cout, nullptr, SC, SH, Cout, HW, Kpad);
    }
  };

  // ---------------- tokenizer ----------------
  // conv0+bn0 (fp32 direct, input identical across T -> once for B)
  prep_bn<<<G(48, 64), 64, 0, stream>>>(P(2), P(3), P(4), P(5), nullptr, SC, SH, 48);
  conv3x3_bn<<<G((size_t)NB * 48 * 9216, 256), 256, 0, stream>>>(
      pix, P(1), SC, SH, RA, NB, 3, 48, 96, 96);
  lif_cf_to_cl<<<G((size_t)NB * 9216 * 48, 256), 256, 0, stream>>>(RA, s0, NB, 48, 9216, 1.f);

  conv_wmma(s0, 96, 96, 48, 96, 6, RC);                       // conv1 -> y1
  lif_f16<<<G(7077888, 256), 256, 0, stream>>>(RC, s1, 7077888, 1.f);

  conv_wmma(s1, 96, 96, 96, 192, 11, RE);                     // conv2 -> y2
  maxpool3s2_cl<<<G((size_t)TBATCH * 2304 * 192, 256), 256, 0, stream>>>(RE, RF, TBATCH, 96, 96, 192);
  lif_f16<<<G(3538944, 256), 256, 0, stream>>>(RF, p2s, 3538944, 1.f);

  conv_wmma(p2s, 48, 48, 192, 384, 16, RC);                   // conv3 -> y3
  maxpool3s2_cl<<<G((size_t)TBATCH * 576 * 384, 256), 256, 0, stream>>>(RC, RF, TBATCH, 48, 48, 384);
  lif_f16<<<G(1769472, 256), 256, 0, stream>>>(RF, p3s, 1769472, 1.f);

  // conv4 writes master x directly (token-major, BN fused)
  {
    int K = 384 * 9, Kpad = K;  // 3456, already mult of 32
    cvt_f16_pad<<<G((size_t)384 * Kpad, 256), 256, 0, stream>>>(P(21), WA, 384, K, Kpad);
    prep_bn<<<G(384, 64), 64, 0, stream>>>(P(22), P(23), P(24), P(25), nullptr, SC, SH, 384);
    for (int img = 0; img < TBATCH; ++img) {
      _Float16* cb = (img & 1) ? COL1 : COL0;
      im2col3x3<<<G((size_t)576 * Kpad, 256), 256, 0, stream>>>(
          p3s + (size_t)img * 576 * 384, cb, 24, 24, 384, Kpad);
      gemm_tok<<<dim3(4, 18, 1), GBLK, 0, stream>>>(
          WA, cb, RX + (size_t)img * 576 * 384, nullptr, SC, SH, 384, 576, Kpad);
    }
  }

  // ---------------- transformer blocks ----------------
  for (int j = 0; j < 4; ++j) {
    int bb = 26 + j * 33;

    lif_f16<<<G(1769472, 256), 256, 0, stream>>>(RX, xl_h, 1769472, 1.f);

    const int tagoff[3] = {0, 5, 10};
    _Float16* sdst[3] = {q_h, k_h, v_h};
    for (int t = 0; t < 3; ++t) {
      int w = bb + tagoff[t];
      cvt_f16_pad<<<G(147456, 256), 256, 0, stream>>>(P(w), WA, 384, 384, 384);
      prep_bn<<<G(384, 64), 64, 0, stream>>>(P(w + 1), P(w + 2), P(w + 3), P(w + 4),
                                             nullptr, SC, SH, 384);
      gemm_tok<<<dim3(4, 18, TBATCH), GBLK, 0, stream>>>(
          WA, xl_h, z_tmp, nullptr, SC, SH, EMB, NTOK, EMB);
      lif_f16<<<G(1769472, 256), 256, 0, stream>>>(z_tmp, sdst[t], 1769472, 1.f);
    }
    v_to_headT<<<G(7077888, 256), 256, 0, stream>>>(v_h, vT_h, TBATCH, NTOK, EMB);

    attn_fused<<<dim3(36, 1, TBATCH * HEADS), 32, 0, stream>>>(q_h, k_h, vT_h, o_f,
                                                               NTOK, EMB);
    lif_f16<<<G(1769472, 256), 256, 0, stream>>>(o_f, o_h, 1769472, 0.5f);  // vth=0.5

    cvt_f16_pad<<<G(147456, 256), 256, 0, stream>>>(P(bb + 15), WA, 384, 384, 384);
    prep_bn<<<G(384, 64), 64, 0, stream>>>(P(bb + 17), P(bb + 18), P(bb + 19), P(bb + 20),
                                           P(bb + 16), SC, SH, 384);
    gemm_tok<<<dim3(4, 18, TBATCH), GBLK, 0, stream>>>(
        WA, o_h, RX, RX, SC, SH, EMB, NTOK, EMB);

    lif_f16<<<G(1769472, 256), 256, 0, stream>>>(RX, xl_h, 1769472, 1.f);
    cvt_f16_pad<<<G(589824, 256), 256, 0, stream>>>(P(bb + 21), WA, 1536, 384, 384);
    prep_bn<<<G(1536, 64), 64, 0, stream>>>(P(bb + 23), P(bb + 24), P(bb + 25), P(bb + 26),
                                            P(bb + 22), SC, SH, 1536);
    gemm_tok<<<dim3(16, 18, TBATCH), GBLK, 0, stream>>>(
        WA, xl_h, z_tmp, nullptr, SC, SH, HIDDEN, NTOK, EMB);
    lif_f16<<<G(7077888, 256), 256, 0, stream>>>(z_tmp, h_big, 7077888, 1.f);
    cvt_f16_pad<<<G(589824, 256), 256, 0, stream>>>(P(bb + 27), WA, 384, 1536, 1536);
    prep_bn<<<G(384, 64), 64, 0, stream>>>(P(bb + 29), P(bb + 30), P(bb + 31), P(bb + 32),
                                           P(bb + 28), SC, SH, 384);
    gemm_tok<<<dim3(4, 18, TBATCH), GBLK, 0, stream>>>(
        WA, h_big, RX, RX, SC, SH, EMB, NTOK, HIDDEN);
  }

  // ---------------- head ----------------
  head_reduce<<<G(3072, 256), 256, 0, stream>>>(RX, P(158), FZ, NB, NTOK, EMB);
  final_proj<<<G(4096, 256), 256, 0, stream>>>(FZ, P(159), P(160), (float*)d_out,
                                               NB, 512, EMB);
}
